// AttentionBlock_71923522339255
// MI455X (gfx1250) — matrix operand order repaired and checked by reading, not compiled, not run on hardware
//
#include <hip/hip_runtime.h>

#define BATCH 16
#define CCH   512
#define NPIX  1024
#define HEADS 4
#define HD    128

typedef __attribute__((ext_vector_type(16))) _Float16 v16h;
typedef __attribute__((ext_vector_type(8)))  float    v8f;
typedef __attribute__((ext_vector_type(4)))  unsigned int v4u;
typedef __attribute__((ext_vector_type(8)))  int      v8i;
typedef __attribute__((ext_vector_type(4)))  int      v4i;

union FragH16 { v16h h; v4u u[2]; };
union Half8   { v4u u; _Float16 e[8]; };

#if defined(__AMDGCN__) && defined(__gfx1250__) && \
    __has_builtin(__builtin_amdgcn_tensor_load_to_lds)
#define HAVE_TDM 1
#endif

__device__ __forceinline__ void wait_tensorcnt0() {
#if defined(__AMDGCN__) && __has_builtin(__builtin_amdgcn_s_wait_tensorcnt)
  __builtin_amdgcn_s_wait_tensorcnt(0);
#elif defined(__AMDGCN__) && defined(__gfx1250__)
  asm volatile("s_wait_tensorcnt 0" ::: "memory");
#endif
}

#ifdef HAVE_TDM
// Issue a 2D TDM tile load: tile_d0 x tile_d1 elements (2 bytes each), row
// stride = stride0 elements, into LDS at byte offset lds_off.
// D# packing per CDNA5 ISA 8.3/8.4 (count=1, type=2 "image", data_size=2B).
__device__ __forceinline__ void tdm_load_2d(unsigned lds_off, const void* gaddr,
                                            unsigned tile_d0, unsigned tile_d1,
                                            unsigned tensor_d0, unsigned tensor_d1,
                                            unsigned long stride0) {
  unsigned long ga = (unsigned long)(uintptr_t)gaddr;
  v4u g0;
  g0[0] = 1u;                                         // count=1, user D#
  g0[1] = lds_off;                                    // lds_addr (bytes)
  g0[2] = (unsigned)(ga & 0xffffffffu);               // global_addr[31:0]
  g0[3] = (unsigned)((ga >> 32) & 0x01ffffffu) | (2u << 30);  // addr[56:32], type=2
  v8i g1;
  g1[0] = (int)(1u << 16);                            // data_size=1 (2 bytes)
  g1[1] = (int)((tensor_d0 & 0xffffu) << 16);         // tensor_dim0[15:0]
  g1[2] = (int)((tensor_d0 >> 16) | ((tensor_d1 & 0xffffu) << 16));
  g1[3] = (int)((tensor_d1 >> 16) | ((tile_d0 & 0xffffu) << 16));
  g1[4] = (int)(tile_d1 & 0xffffu);                   // tile_dim1; tile_dim2=0
  g1[5] = (int)(unsigned)(stride0 & 0xffffffffu);     // tensor_dim0_stride[31:0]
  g1[6] = (int)((unsigned)(stride0 >> 32) & 0xffffu); // stride[47:32]; dim1_stride=0
  g1[7] = 0;
  v4i z4 = {0, 0, 0, 0};
#if __has_include(<hip/amd_detail/amd_gfx1250_TDM.h>)
  v8i z8 = {0, 0, 0, 0, 0, 0, 0, 0};
  __builtin_amdgcn_tensor_load_to_lds(g0, g1, z4, z4, z8, 0);
#else
  __builtin_amdgcn_tensor_load_to_lds(g0, g1, z4, z4, 0);
#endif
}
#endif

// A-matrix 16x32 f16 fragment from row-major src (rows = M, ld = K-stride).
__device__ __forceinline__ v16h loadA16(const _Float16* base, int ld, int lane) {
  int lr = lane & 15, hi = lane >> 4;
  FragH16 f;
  const _Float16* p = base + (size_t)lr * ld + hi * 8;
  f.u[0] = *(const v4u*)(p);
  f.u[1] = *(const v4u*)(p + 16);
  return f.h;
}

// B-matrix 32x16 f16 fragment from src stored [n][k] (k contiguous).
__device__ __forceinline__ v16h loadB16(const _Float16* base, int ld, int lane) {
  int lr = lane & 15, hi = lane >> 4;
  FragH16 f;
  const _Float16* p = base + (size_t)lr * ld + hi * 8;
  f.u[0] = *(const v4u*)(p);
  f.u[1] = *(const v4u*)(p + 16);
  return f.h;
}

__device__ __forceinline__ v8f wmma16(v16h a, v16h b, v8f c) {
  return __builtin_amdgcn_wmma_f32_16x16x32_f16(false, a, false, b, (short)0, c,
                                                false, false);
}

// ---------------- weight fp32 -> f16 ----------------
__global__ void __launch_bounds__(256) cvt_kernel(const float* src, _Float16* dst, int n) {
  int i = blockIdx.x * 256 + threadIdx.x;
  if (i < n) dst[i] = (_Float16)src[i];
}

// ---------------- GroupNorm -> h16 [b][n][c] (token-major f16) ----------------
__global__ void __launch_bounds__(256) gn_kernel(const float* x, const float* gw,
                                                 const float* gb, _Float16* h16) {
  __shared__ float red[512];
  int b = blockIdx.x >> 5;
  int g = blockIdx.x & 31;
  const float* xg = x + ((size_t)b * CCH + g * 16) * NPIX;
  float s = 0.f, s2 = 0.f;
  for (int e = threadIdx.x; e < 16 * NPIX; e += 256) {
    float v = xg[e];
    s += v; s2 += v * v;
  }
  red[threadIdx.x] = s;
  red[256 + threadIdx.x] = s2;
  __syncthreads();
  for (int o = 128; o > 0; o >>= 1) {
    if ((int)threadIdx.x < o) {
      red[threadIdx.x]       += red[threadIdx.x + o];
      red[256 + threadIdx.x] += red[256 + threadIdx.x + o];
    }
    __syncthreads();
  }
  float mean = red[0] * (1.f / 16384.f);
  float var  = red[256] * (1.f / 16384.f) - mean * mean;
  float inv  = rsqrtf(var + 1e-5f);
  for (int e = threadIdx.x; e < 16 * NPIX; e += 256) {
    int cl = e >> 10;
    int n  = e & 1023;
    int c  = g * 16 + cl;
    float v = (xg[e] - mean) * inv * gw[c] + gb[c];
    h16[((size_t)b * NPIX + n) * CCH + c] = (_Float16)v;
  }
}

// ---------------- QKV GEMM: [1536x512] x [512x1024] per batch ----------------
__global__ void __launch_bounds__(256) qkv_kernel(const _Float16* w16, const float* bias,
                                                  const _Float16* h16, _Float16* q16,
                                                  _Float16* k16, _Float16* vt16) {
  int lane = threadIdx.x & 31;
  int lr = lane & 15, hi = lane >> 4;
  int wid = blockIdx.x * 8 + (threadIdx.x >> 5);
  int b = wid / 768;
  int t = wid - b * 768;              // 48 M-tiles(32) x 16 N-tiles(64)
  int m0 = (t >> 4) * 32;
  int n0 = (t & 15) * 64;
  const _Float16* hb = h16 + (size_t)b * NPIX * CCH;
  v8f acc[2][4];
#pragma unroll
  for (int mi = 0; mi < 2; ++mi)
#pragma unroll
    for (int tt = 0; tt < 4; ++tt) acc[mi][tt] = (v8f)(0.f);
  for (int k0 = 0; k0 < CCH; k0 += 32) {
    v16h a0 = loadA16(w16 + (size_t)m0 * CCH + k0, CCH, lane);
    v16h a1 = loadA16(w16 + (size_t)(m0 + 16) * CCH + k0, CCH, lane);
    __builtin_prefetch(w16 + (size_t)m0 * CCH + k0 + 64, 0, 0);
#pragma unroll
    for (int tt = 0; tt < 4; ++tt) {
      v16h bf = loadB16(hb + (size_t)(n0 + tt * 16) * CCH + k0, CCH, lane);
      acc[0][tt] = wmma16(a0, bf, acc[0][tt]);
      acc[1][tt] = wmma16(a1, bf, acc[1][tt]);
    }
  }
  const float scale = 0.08838834764831845f;  // 1/sqrt(128)
#pragma unroll
  for (int mi = 0; mi < 2; ++mi) {
    int m0i = m0 + mi * 16;
    float bv[8];
#pragma unroll
    for (int r = 0; r < 8; ++r) bv[r] = bias[m0i + hi * 8 + r];
    int s    = m0i >> 9;               // 0=q 1=k 2=v
    int rch  = m0i & 511;
    int head = rch >> 7;
    int d0   = (rch & 127) + hi * 8;
#pragma unroll
    for (int tt = 0; tt < 4; ++tt) {
      int n = n0 + tt * 16 + lr;
      size_t tokbase = (((size_t)b * HEADS + head) * NPIX + n) * HD;
      if (s == 0) {
        Half8 p;
#pragma unroll
        for (int r = 0; r < 8; ++r) p.e[r] = (_Float16)((acc[mi][tt][r] + bv[r]) * scale);
        *(v4u*)(q16 + tokbase + d0) = p.u;
      } else if (s == 1) {
        Half8 p;
#pragma unroll
        for (int r = 0; r < 8; ++r) p.e[r] = (_Float16)(acc[mi][tt][r] + bv[r]);
        *(v4u*)(k16 + tokbase + d0) = p.u;
      } else {
        size_t vb = ((size_t)b * HEADS + head) * HD;
#pragma unroll
        for (int r = 0; r < 8; ++r)
          vt16[(vb + d0 + r) * NPIX + n] = (_Float16)(acc[mi][tt][r] + bv[r]);
      }
    }
  }
}

// ---------------- Flash attention per (b, head, 128-query block) ----------------
// Double-buffered K/V LDS staging; next block DMA'd by the Tensor Data Mover
// (fallback: register-staged cooperative copy) while WMMAs run on the current one.
__global__ void __launch_bounds__(256) attn_kernel(const _Float16* q16, const _Float16* k16,
                                                   const _Float16* vt16, _Float16* o16) {
  __shared__ _Float16 ldsK[2 * 32 * 128];  // 16KB [buf][j_local][d]
  __shared__ _Float16 ldsV[2 * 128 * 32];  // 16KB [buf][d][j_local]
  __shared__ _Float16 ldsO[8 * 16 * 128];  // 32KB O tiles; 512 halves/wave reused as P slab
  int tid = threadIdx.x;
  int lane = tid & 31;
  int w = tid >> 5;
  int lr = lane & 15, hi = lane >> 4;
  int qb   = blockIdx.x & 7;
  int head = (blockIdx.x >> 3) & 3;
  int b    = blockIdx.x >> 5;
  const size_t hdoff = ((size_t)b * HEADS + head) * (size_t)NPIX * HD;
  const _Float16* qp = q16 + hdoff;
  const _Float16* kp = k16 + hdoff;
  const _Float16* vp = vt16 + hdoff;
  int i0 = qb * 128 + w * 16;              // 16 query rows per wave
  _Float16* lp = ldsO + w * 2048;          // per-wave P staging slab (private)

  v16h aq[4];                              // Q 16x128 as 4 A-fragments
#pragma unroll
  for (int c = 0; c < 4; ++c) aq[c] = loadA16(qp + (size_t)i0 * HD + c * 32, HD, lane);

  v8f o[8];
#pragma unroll
  for (int f = 0; f < 8; ++f) o[f] = (v8f)(0.f);
  float rmax[8], rsum[8];
#pragma unroll
  for (int r = 0; r < 8; ++r) { rmax[r] = -1e30f; rsum[r] = 0.f; }

  int kr = tid >> 3, ks = tid & 7;         // K stage: row 0..31, 16-half seg 0..7
  int vr = tid >> 1, vs = tid & 1;         // V stage: row 0..127, 16-half seg 0..1

  // ---- prologue: stage key-block 0 into buffer 0 ----
#ifdef HAVE_TDM
  if (w == 0) {
    tdm_load_2d((unsigned)(uintptr_t)(ldsK), kp, HD, 32, HD, NPIX, HD);
    tdm_load_2d((unsigned)(uintptr_t)(ldsV), vp, 32, HD, NPIX, HD, NPIX);
    wait_tensorcnt0();
  }
#else
  {
    const _Float16* gk = kp + (size_t)kr * HD + ks * 16;
    *(v4u*)(ldsK + kr * 128 + ks * 16)     = *(const v4u*)(gk);
    *(v4u*)(ldsK + kr * 128 + ks * 16 + 8) = *(const v4u*)(gk + 8);
    const _Float16* gv = vp + (size_t)vr * NPIX + vs * 16;
    *(v4u*)(ldsV + vr * 32 + vs * 16)      = *(const v4u*)(gv);
    *(v4u*)(ldsV + vr * 32 + vs * 16 + 8)  = *(const v4u*)(gv + 8);
  }
#endif
  __syncthreads();

  for (int jt = 0; jt < 32; ++jt) {
    int cur = jt & 1, nxt = cur ^ 1;
    int j0 = jt * 32;
    const _Float16* kb = ldsK + cur * 4096;
    const _Float16* vb = ldsV + cur * 4096;
    // ---- kick off next block's staging (overlaps with WMMAs below) ----
#ifdef HAVE_TDM
    if (jt + 1 < 32 && w == 0) {
      tdm_load_2d((unsigned)(uintptr_t)(ldsK + nxt * 4096),
                  kp + (size_t)(j0 + 32) * HD, HD, 32, HD, NPIX, HD);
      tdm_load_2d((unsigned)(uintptr_t)(ldsV + nxt * 4096),
                  vp + (j0 + 32), 32, HD, NPIX, HD, NPIX);
    }
#else
    v4u pk0 = {}, pk1 = {}, pv0 = {}, pv1 = {};
    if (jt + 1 < 32) {
      const _Float16* gk = kp + (size_t)(j0 + 32 + kr) * HD + ks * 16;
      pk0 = *(const v4u*)(gk);
      pk1 = *(const v4u*)(gk + 8);
      const _Float16* gv = vp + (size_t)vr * NPIX + (j0 + 32) + vs * 16;
      pv0 = *(const v4u*)(gv);
      pv1 = *(const v4u*)(gv + 8);
    }
#endif
    // ---- S = Q K^T for 32 keys: two 16x16 f32 tiles, B fragments from LDS ----
    v8f s0 = (v8f)(0.f), s1 = (v8f)(0.f);
#pragma unroll
    for (int c = 0; c < 4; ++c) {
      v16h bk0 = loadB16(kb + c * 32, HD, lane);
      v16h bk1 = loadB16(kb + 16 * 128 + c * 32, HD, lane);
      s0 = wmma16(aq[c], bk0, s0);
      s1 = wmma16(aq[c], bk1, s1);
    }
    // ---- online softmax; rows live in 16-lane groups -> xor masks 1,2,4,8 ----
    float p0[8], p1[8], corr[8];
#pragma unroll
    for (int r = 0; r < 8; ++r) {
      float m = fmaxf(s0[r], s1[r]);
      m = fmaxf(m, __shfl_xor(m, 1, 32));
      m = fmaxf(m, __shfl_xor(m, 2, 32));
      m = fmaxf(m, __shfl_xor(m, 4, 32));
      m = fmaxf(m, __shfl_xor(m, 8, 32));
      float nm = fmaxf(rmax[r], m);
      corr[r] = __expf(rmax[r] - nm);
      rmax[r] = nm;
      p0[r] = __expf(s0[r] - nm);
      p1[r] = __expf(s1[r] - nm);
      float ts = p0[r] + p1[r];
      ts += __shfl_xor(ts, 1, 32);
      ts += __shfl_xor(ts, 2, 32);
      ts += __shfl_xor(ts, 4, 32);
      ts += __shfl_xor(ts, 8, 32);
      rsum[r] = rsum[r] * corr[r] + ts;
    }
#pragma unroll
    for (int f = 0; f < 8; ++f)
#pragma unroll
      for (int r = 0; r < 8; ++r) o[f][r] *= corr[r];
    // ---- transpose P (D-layout) -> A-layout through private slab ----
#pragma unroll
    for (int r = 0; r < 8; ++r) {
      lp[(hi * 8 + r) * 32 + lr]      = (_Float16)p0[r];
      lp[(hi * 8 + r) * 32 + 16 + lr] = (_Float16)p1[r];
    }
    asm volatile("s_wait_dscnt 0" ::: "memory");  // same-wave DS store->load
    FragH16 ap;
    ap.u[0] = *(const v4u*)(lp + lr * 32 + hi * 8);
    ap.u[1] = *(const v4u*)(lp + lr * 32 + 16 + hi * 8);
    // ---- O += P (16x32) x V (32x128), V fragments from LDS [d][j] ----
#pragma unroll
    for (int f = 0; f < 8; ++f) {
      v16h bv = loadB16(vb + f * 16 * 32, 32, lane);
      o[f] = wmma16(ap.h, bv, o[f]);
    }
    // ---- publish next buffer ----
#ifdef HAVE_TDM
    if (jt + 1 < 32 && w == 0) wait_tensorcnt0();
#else
    if (jt + 1 < 32) {
      _Float16* dk = ldsK + nxt * 4096 + kr * 128 + ks * 16;
      *(v4u*)(dk)     = pk0;
      *(v4u*)(dk + 8) = pk1;
      _Float16* dv = ldsV + nxt * 4096 + vr * 32 + vs * 16;
      *(v4u*)(dv)     = pv0;
      *(v4u*)(dv + 8) = pv1;
    }
#endif
    __syncthreads();
  }
  // ---- normalize; park O tile in LDS; coalesced b128 writeback ----
  float inv[8];
#pragma unroll
  for (int r = 0; r < 8; ++r) inv[r] = 1.f / rsum[r];
#pragma unroll
  for (int f = 0; f < 8; ++f)
#pragma unroll
    for (int r = 0; r < 8; ++r)
      ldsO[w * 2048 + (hi * 8 + r) * 128 + f * 16 + lr] = (_Float16)(o[f][r] * inv[r]);
  __syncthreads();
  {
    int row = tid >> 1, seg = tid & 1;     // row 0..127, 64-half segment
    const _Float16* src = ldsO + row * 128 + seg * 64;
    _Float16* dst = o16 + ((size_t)b * NPIX + qb * 128 + row) * CCH + head * HD + seg * 64;
#pragma unroll
    for (int q = 0; q < 8; ++q)
      *(v4u*)(dst + q * 8) = *(const v4u*)(src + q * 8);
  }
}

// ---------------- proj GEMM + bias + residual -> fp32 out [b][c][hw] ----------------
__global__ void __launch_bounds__(256) proj_kernel(const _Float16* w16, const float* bias,
                                                   const _Float16* o16, const float* x,
                                                   float* out) {
  int lane = threadIdx.x & 31;
  int lr = lane & 15, hi = lane >> 4;
  int wid = blockIdx.x * 8 + (threadIdx.x >> 5);
  int b = wid >> 8;
  int t = wid & 255;                   // 16 M-tiles(32) x 16 N-tiles(64)
  int m0 = (t >> 4) * 32;
  int n0 = (t & 15) * 64;
  const _Float16* ob = o16 + (size_t)b * NPIX * CCH;
  v8f acc[2][4];
#pragma unroll
  for (int mi = 0; mi < 2; ++mi)
#pragma unroll
    for (int tt = 0; tt < 4; ++tt) acc[mi][tt] = (v8f)(0.f);
  for (int k0 = 0; k0 < CCH; k0 += 32) {
    v16h a0 = loadA16(w16 + (size_t)m0 * CCH + k0, CCH, lane);
    v16h a1 = loadA16(w16 + (size_t)(m0 + 16) * CCH + k0, CCH, lane);
    __builtin_prefetch(w16 + (size_t)m0 * CCH + k0 + 64, 0, 0);
#pragma unroll
    for (int tt = 0; tt < 4; ++tt) {
      v16h bf = loadB16(ob + (size_t)(n0 + tt * 16) * CCH + k0, CCH, lane);
      acc[0][tt] = wmma16(a0, bf, acc[0][tt]);
      acc[1][tt] = wmma16(a1, bf, acc[1][tt]);
    }
  }
#pragma unroll
  for (int mi = 0; mi < 2; ++mi) {
    int m0i = m0 + mi * 16;
    float bv[8];
#pragma unroll
    for (int r = 0; r < 8; ++r) bv[r] = bias[m0i + hi * 8 + r];
#pragma unroll
    for (int tt = 0; tt < 4; ++tt) {
      int n = n0 + tt * 16 + lr;
#pragma unroll
      for (int r = 0; r < 8; ++r) {
        int m = m0i + hi * 8 + r;
        size_t idx = ((size_t)b * CCH + m) * NPIX + n;
        out[idx] = acc[mi][tt][r] + bv[r] + x[idx];
      }
    }
  }
}

extern "C" void kernel_launch(void* const* d_in, const int* in_sizes, int n_in,
                              void* d_out, int out_size, void* d_ws, size_t ws_size,
                              hipStream_t stream) {
  const float* x      = (const float*)d_in[0];
  const float* norm_w = (const float*)d_in[1];
  const float* norm_b = (const float*)d_in[2];
  const float* qkv_w  = (const float*)d_in[3];
  const float* qkv_b  = (const float*)d_in[4];
  const float* proj_w = (const float*)d_in[5];
  const float* proj_b = (const float*)d_in[6];
  float* out = (float*)d_out;

  char* ws = (char*)d_ws;
  _Float16* wqkv16  = (_Float16*)(ws + 0x0000000);  // 1536*512*2 = 0x180000
  _Float16* wproj16 = (_Float16*)(ws + 0x0180000);  // 512*512*2  = 0x080000
  _Float16* h16     = (_Float16*)(ws + 0x0200000);  // 16MB [b][n][c]
  _Float16* q16     = (_Float16*)(ws + 0x1200000);  // 16MB [b][h][n][d]
  _Float16* k16     = (_Float16*)(ws + 0x2200000);  // 16MB [b][h][n][d]
  _Float16* vt16    = (_Float16*)(ws + 0x3200000);  // 16MB [b][h][d][n]
  _Float16* o16     = (_Float16*)(ws + 0x4200000);  // 16MB [b][n][c]

  cvt_kernel<<<3072, 256, 0, stream>>>(qkv_w, wqkv16, 1536 * 512);
  cvt_kernel<<<1024, 256, 0, stream>>>(proj_w, wproj16, 512 * 512);
  gn_kernel<<<512, 256, 0, stream>>>(x, norm_w, norm_b, h16);
  qkv_kernel<<<1536, 256, 0, stream>>>(wqkv16, qkv_b, h16, q16, k16, vt16);
  attn_kernel<<<512, 256, 0, stream>>>(q16, k16, vt16, o16);
  proj_kernel<<<512, 256, 0, stream>>>(wproj16, proj_b, o16, x, out);
}
